// EGC_4398046511486
// MI455X (gfx1250) — matrix operand order repaired and checked by reading, not compiled, not run on hardware
//
#include <hip/hip_runtime.h>
#include <cfloat>

typedef __attribute__((ext_vector_type(16))) __bf16 v16bf;
typedef __attribute__((ext_vector_type(8)))  __bf16 v8bf;
typedef __attribute__((ext_vector_type(8)))  float  v8f;

#define TB 256

// ---------------------------------------------------------------- utilities
__global__ void fillk(float* __restrict__ p, long n, float v) {
  long i = (long)blockIdx.x * blockDim.x + threadIdx.x;
  if (i < n) p[i] = v;
}

__global__ void deg_kernel(const int* __restrict__ row, float* __restrict__ deg, int E) {
  int e = blockIdx.x * blockDim.x + threadIdx.x;
  if (e < E) atomicAdd(&deg[row[e]], 1.0f);
}

__global__ void dinv_kernel(const float* __restrict__ deg, float* __restrict__ dinv, int N) {
  int n = blockIdx.x * blockDim.x + threadIdx.x;
  if (n < N) {
    float d = deg[n];
    dinv[n] = d > 0.0f ? rsqrtf(fmaxf(d, 1.0f)) : 0.0f;
  }
}

__global__ void norm_kernel(const int* __restrict__ row, const int* __restrict__ col,
                            const float* __restrict__ dinv, float* __restrict__ nrm, int E) {
  int e = blockIdx.x * blockDim.x + threadIdx.x;
  if (e < E) nrm[e] = dinv[row[e]] * dinv[col[e]];
}

// f32 -> bf16 activation staging (hot GEMM then reads half the bytes)
__global__ void cvt_bf16(const float* __restrict__ s, __bf16* __restrict__ d, long n) {
  long i = (long)blockIdx.x * blockDim.x + threadIdx.x;
  if (i < n) d[i] = (__bf16)s[i];
}

// W[K,Nc] f32 -> WT[Nc,K] bf16 (makes B-fragment loads contiguous b128s)
__global__ void transpose_cvt(const float* __restrict__ W, __bf16* __restrict__ WT,
                              int K, int Nc) {
  int idx = blockIdx.x * blockDim.x + threadIdx.x;
  if (idx < K * Nc) {
    int k = idx / Nc;
    int n = idx - k * Nc;
    WT[(size_t)n * K + k] = (__bf16)W[idx];
  }
}

// ---------------------------------------------------- WMMA bf16 GEMM (f32 acc)
// Y[M,Nc] = A[M,K](bf16) * WT[Nc,K](bf16, pre-transposed) (+ bias[Nc])
// Block: 256 threads = 8 waves; each wave a 32x16 tile (2 accumulators sharing
// one B fragment) -> 2 v_wmma per 32-K step. K % 32 == 0, Nc % 16 == 0.
// Fragment maps used here (A and B share one contiguous k order):
//   A 16x32:  lane(l16,hi) elem e -> K = e + 16*hi
//             => two contiguous 8-elem (16B) loads per tile.
//   B 32x16:  lane holds col N=l16, elem e -> K = e + 16*hi
//             => one contiguous 16-elem (32B) load from WT.
//   C/D f32:  VGPR v -> row M = v + 8*hi.
__global__ __launch_bounds__(256) void gemm_bf16_wmma(
    const __bf16* __restrict__ A, const __bf16* __restrict__ WT,
    const float* __restrict__ bias, float* __restrict__ Y,
    int M, int K, int Nc) {
  const int lane = threadIdx.x & 31;
  const int wave = threadIdx.x >> 5;
  const int i0 = blockIdx.x * 256 + wave * 32;
  const int j0 = blockIdx.y * 16;
  const int l16 = lane & 15;
  const int hi = (lane & 16) ? 1 : 0;

  int m0 = i0 + l16;
  int m1 = i0 + 16 + l16;
  int m0c = m0 < M ? m0 : M - 1;                 // clamp OOB rows (stores guarded)
  int m1c = m1 < M ? m1 : M - 1;
  const __bf16* __restrict__ a0p = A + (size_t)m0c * K + 16 * hi;
  const __bf16* __restrict__ a1p = A + (size_t)m1c * K + 16 * hi;
  const __bf16* __restrict__ bp  = WT + (size_t)(j0 + l16) * K + 16 * hi;

  v8f acc0 = {}, acc1 = {};
  for (int k = 0; k < K; k += 32) {
    v8bf a0lo = *(const v8bf*)(a0p + k);
    v8bf a0hi = *(const v8bf*)(a0p + k + 8);
    v8bf a1lo = *(const v8bf*)(a1p + k);
    v8bf a1hi = *(const v8bf*)(a1p + k + 8);
    v16bf b = *(const v16bf*)(bp + k);
    v16bf a0 = __builtin_shufflevector(a0lo, a0hi, 0, 1, 2, 3, 4, 5, 6, 7,
                                       8, 9, 10, 11, 12, 13, 14, 15);
    v16bf a1 = __builtin_shufflevector(a1lo, a1hi, 0, 1, 2, 3, 4, 5, 6, 7,
                                       8, 9, 10, 11, 12, 13, 14, 15);
    acc0 = __builtin_amdgcn_wmma_f32_16x16x32_bf16(false, a0, false, b,
                                                   (short)0, acc0, false, false);
    acc1 = __builtin_amdgcn_wmma_f32_16x16x32_bf16(false, a1, false, b,
                                                   (short)0, acc1, false, false);
  }

  float bv = bias ? bias[j0 + l16] : 0.0f;
  if (i0 + 31 < M) {                              // fast path: full tile
    float* y0 = Y + (size_t)(i0 + 8 * hi) * Nc + j0 + l16;
#pragma unroll
    for (int v = 0; v < 8; ++v) {
      y0[(size_t)v * Nc] = acc0[v] + bv;
      y0[(size_t)(v + 16) * Nc] = acc1[v] + bv;
    }
  } else {
#pragma unroll
    for (int v = 0; v < 8; ++v) {
      int mm0 = i0 + v + 8 * hi;
      int mm1 = mm0 + 16;
      if (mm0 < M) Y[(size_t)mm0 * Nc + j0 + l16] = acc0[v] + bv;
      if (mm1 < M) Y[(size_t)mm1 * Nc + j0 + l16] = acc1[v] + bv;
    }
  }
}

// ------------------------------------------------------------- aggregation
__device__ __forceinline__ void atomicMaxF(float* addr, float v) {
  // IEEE-754 monotone bit-pattern trick; valid for mixed signs.
  if (v >= 0.0f) atomicMax((int*)addr, __float_as_int(v));
  else           atomicMin((unsigned int*)addr, __float_as_uint(v));
}

__global__ void aggregate_kernel(const float* __restrict__ bas,
                                 const int* __restrict__ row, const int* __restrict__ col,
                                 const float* __restrict__ nrm,
                                 float* __restrict__ sym, float* __restrict__ sums,
                                 float* __restrict__ mx, int E, int NBF) {
  long idx = (long)blockIdx.x * blockDim.x + threadIdx.x;
  int per = NBF >> 2;                  // float4 groups per edge
  long tot = (long)E * per;
  if (idx >= tot) return;
  int e = (int)(idx / per);
  int g = ((int)(idx - (long)e * per)) << 2;
  int r = row[e], c = col[e];
  float w = nrm[e];
  const float4 v = *(const float4*)(bas + (size_t)c * NBF + g);
  size_t o = (size_t)r * NBF + g;
  atomicAdd(sym + o + 0, v.x * w); atomicAdd(sym + o + 1, v.y * w);
  atomicAdd(sym + o + 2, v.z * w); atomicAdd(sym + o + 3, v.w * w);
  atomicAdd(sums + o + 0, v.x); atomicAdd(sums + o + 1, v.y);
  atomicAdd(sums + o + 2, v.z); atomicAdd(sums + o + 3, v.w);
  atomicMaxF(mx + o + 0, v.x); atomicMaxF(mx + o + 1, v.y);
  atomicMaxF(mx + o + 2, v.z); atomicMaxF(mx + o + 3, v.w);
}

// ------------------------------------------------- per-node mixing (einsum)
// out[n, h*F+f] = b[h*F+f] + sum_{k=0..11} comb[n, h*12+k] * agg[n,k,f]
// agg k = a*NB + j -> aggregator a, basis block j. Vectorized over 4 f's
// (F % 4 == 0, so a group never crosses a head boundary).
__global__ void combine_kernel(const float* __restrict__ sym, const float* __restrict__ sums,
                               const float* __restrict__ mx, const float* __restrict__ comb,
                               const float* __restrict__ bias, const float* __restrict__ deg,
                               float* __restrict__ out, int N, int F, int NBF, int dout,
                               int do_relu) {
  long idx = (long)blockIdx.x * blockDim.x + threadIdx.x;
  int q4 = dout >> 2;
  long tot = (long)N * q4;
  if (idx >= tot) return;
  int n = (int)(idx / q4);
  int r = ((int)(idx - (long)n * q4)) << 2;
  int h = r / F;
  int f = r - h * F;
  float d = deg[n];
  float invd = 1.0f / fmaxf(d, 1.0f);
  float hasd = d > 0.0f ? 1.0f : 0.0f;            // empty segments: max -> 0
  const float* cw = comb + (size_t)n * 96 + h * 12;
  size_t base = (size_t)n * NBF + f;
  float4 acc = *(const float4*)(bias + r);
#pragma unroll
  for (int k = 0; k < 12; ++k) {
    int a = k >> 2;                               // aggregator (NB == 4)
    int j = k & 3;                                // basis block
    const float* src = (a == 0) ? sym : (a == 1) ? sums : mx;
    float s = (a == 0) ? 1.0f : (a == 1) ? invd : hasd;
    float4 av = *(const float4*)(src + base + (size_t)j * F);
    float wk = cw[k] * s;
    acc.x += wk * av.x; acc.y += wk * av.y;
    acc.z += wk * av.z; acc.w += wk * av.w;
  }
  if (do_relu) {
    acc.x = fmaxf(acc.x, 0.0f); acc.y = fmaxf(acc.y, 0.0f);
    acc.z = fmaxf(acc.z, 0.0f); acc.w = fmaxf(acc.w, 0.0f);
  }
  *(float4*)(out + (size_t)n * dout + r) = acc;
}

// ---------------------------------------------------------- log-softmax
__global__ void logsoftmax_kernel(const float* __restrict__ in, float* __restrict__ out, int N) {
  int n = blockIdx.x;
  int lane = threadIdx.x;                         // blockDim.x == 32 (one wave)
  const float* p = in + (size_t)n * 352;
  float mv = -FLT_MAX;
  for (int i = lane; i < 349; i += 32) mv = fmaxf(mv, p[i]);
  for (int o = 16; o > 0; o >>= 1) mv = fmaxf(mv, __shfl_xor(mv, o, 32));
  float s = 0.0f;
  for (int i = lane; i < 349; i += 32) s += expf(p[i] - mv);
  for (int o = 16; o > 0; o >>= 1) s += __shfl_xor(s, o, 32);
  float lse = mv + logf(s);
  for (int i = lane; i < 349; i += 32) out[(size_t)n * 349 + i] = p[i] - lse;
}

// ---------------------------------------------------------------- driver
extern "C" void kernel_launch(void* const* d_in, const int* in_sizes, int n_in,
                              void* d_out, int out_size, void* d_ws, size_t ws_size,
                              hipStream_t stream) {
  const float* x = (const float*)d_in[0];
  const int* ei = (const int*)d_in[1];
  const int N = in_sizes[0] / 128;
  const int E = in_sizes[1] / 2;
  const int* row = ei;
  const int* col = ei + E;

  const float* Wb[3] = {(const float*)d_in[2], (const float*)d_in[6], (const float*)d_in[10]};
  const float* Wc[3] = {(const float*)d_in[3], (const float*)d_in[7], (const float*)d_in[11]};
  const float* bc[3] = {(const float*)d_in[4], (const float*)d_in[8], (const float*)d_in[12]};
  const float* bb[3] = {(const float*)d_in[5], (const float*)d_in[9], (const float*)d_in[13]};

  float* w = (float*)d_ws;
  float* deg  = w;  w += N;
  float* dinv = w;  w += N;
  float* nrm  = w;  w += E;
  float* hbuf = w;  w += (size_t)N * 256;
  float* bas  = w;  w += (size_t)N * 176;
  float* cmb  = w;  w += (size_t)N * 96;
  float* sym  = w;  w += (size_t)N * 176;
  float* sums = w;  w += (size_t)N * 176;
  float* mx   = w;  w += (size_t)N * 176;
  float* fin  = w;  w += (size_t)N * 352;
  __bf16* hb  = (__bf16*)w;                       // N*256 bf16
  __bf16* wt  = hb + (size_t)N * 256;             // max 176*256 bf16

  // degree / symmetric normalization (recomputed every call: deterministic)
  fillk<<<(N + TB - 1) / TB, TB, 0, stream>>>(deg, N, 0.0f);
  deg_kernel<<<(E + TB - 1) / TB, TB, 0, stream>>>(row, deg, E);
  dinv_kernel<<<(N + TB - 1) / TB, TB, 0, stream>>>(deg, dinv, N);
  norm_kernel<<<(E + TB - 1) / TB, TB, 0, stream>>>(row, col, dinv, nrm, E);

  const float* h = x;
  int din = 128;
  for (int l = 0; l < 3; ++l) {
    const int NBF  = (l == 2) ? 176 : 128;   // NB * F
    const int F    = NBF / 4;
    const int dout = (l == 2) ? 352 : 256;

    // stage activations as bf16 once per layer
    long hn = (long)N * din;
    cvt_bf16<<<(hn + TB - 1) / TB, TB, 0, stream>>>(h, hb, hn);

    // bases = h @ Wb
    transpose_cvt<<<(din * NBF + TB - 1) / TB, TB, 0, stream>>>(Wb[l], wt, din, NBF);
    dim3 gb((N + 255) / 256, NBF / 16);
    gemm_bf16_wmma<<<gb, 256, 0, stream>>>(hb, wt, nullptr, bas, N, din, NBF);

    // comb = h @ Wc + bc
    transpose_cvt<<<(din * 96 + TB - 1) / TB, TB, 0, stream>>>(Wc[l], wt, din, 96);
    dim3 gc((N + 255) / 256, 96 / 16);
    gemm_bf16_wmma<<<gc, 256, 0, stream>>>(hb, wt, bc[l], cmb, N, din, 96);

    long nn = (long)N * NBF;
    fillk<<<(nn + TB - 1) / TB, TB, 0, stream>>>(sym,  nn, 0.0f);
    fillk<<<(nn + TB - 1) / TB, TB, 0, stream>>>(sums, nn, 0.0f);
    fillk<<<(nn + TB - 1) / TB, TB, 0, stream>>>(mx,   nn, -FLT_MAX);

    long tot = (long)E * (NBF / 4);
    aggregate_kernel<<<(int)((tot + TB - 1) / TB), TB, 0, stream>>>(
        bas, row, col, nrm, sym, sums, mx, E, NBF);

    float* dst = (l == 2) ? fin : hbuf;
    long ct = (long)N * (dout / 4);
    combine_kernel<<<(int)((ct + TB - 1) / TB), TB, 0, stream>>>(
        sym, sums, mx, cmb, bb[l], deg, dst, N, F, NBF, dout, (l < 2) ? 1 : 0);

    h = hbuf;
    din = dout;
  }

  logsoftmax_kernel<<<N, 32, 0, stream>>>(fin, (float*)d_out, N);
}
